// RNNDecoder_34497177322207
// MI455X (gfx1250) — compile-verified, matrix operand order checked by reading, not run on hardware
//
#include <hip/hip_runtime.h>

#define EMBED   512
#define HIDDEN  512
#define OUTD    100
#define BATCH   500
#define TSTEPS  256
#define MPAD    512               // batch padded to multiple of 16
#define KCOMB   1024              // E + H
#define NCOMB   2048              // 4*H
#define NT_COMB 128               // NCOMB/16
#define KT_COMB 32                // KCOMB/32
#define NPAD_FC 112               // OUTD padded to multiple of 16
#define NT_FC   7
#define KT_FC   16                // HIDDEN/32

#define CELL_BLOCKS 128           // 1024 waves for the cell GEMM
#define FC_BLOCKS   28            // 224 waves for the FC GEMM

typedef __bf16 bf16;
typedef __attribute__((ext_vector_type(16))) __bf16 v16bf;
typedef __attribute__((ext_vector_type(8)))  __bf16 bf16x8;
typedef __attribute__((ext_vector_type(8)))  float  v8f;

// ---------------------------------------------------------------------------
// 16-bit WMMA K mapping (CDNA5 ISA, "16-bit A-Matrix 16x32"):
//   VGPR 0..3 hold K={0..7} (lanes 0-15) / K={8..15} (lanes 16-31),
//   VGPR 4..7 hold K={16..23} / K={24..31}.  Element e = 2*v + p.
// ---------------------------------------------------------------------------
__device__ __host__ inline int kmap(int v, int half, int p) {
    return ((v < 4) ? 0 : 16) + (half ? 8 : 0) + 2 * (v & 3) + p;
}

__device__ inline float sigmoidf(float x) { return 1.0f / (1.0f + __expf(-x)); }

// Build an A fragment (16x32 bf16) from a row-major fp32 row (this lane's row).
__device__ inline v16bf load_a_f32(const float* __restrict__ row, int kbase, int half) {
    const float* p0 = row + kbase + (half ? 8 : 0);
    float4 a0 = *(const float4*)(p0 + 0);
    float4 a1 = *(const float4*)(p0 + 4);
    float4 a2 = *(const float4*)(p0 + 16);
    float4 a3 = *(const float4*)(p0 + 20);
    v16bf r;
    r[0]  = (bf16)a0.x; r[1]  = (bf16)a0.y; r[2]  = (bf16)a0.z; r[3]  = (bf16)a0.w;
    r[4]  = (bf16)a1.x; r[5]  = (bf16)a1.y; r[6]  = (bf16)a1.z; r[7]  = (bf16)a1.w;
    r[8]  = (bf16)a2.x; r[9]  = (bf16)a2.y; r[10] = (bf16)a2.z; r[11] = (bf16)a2.w;
    r[12] = (bf16)a3.x; r[13] = (bf16)a3.y; r[14] = (bf16)a3.z; r[15] = (bf16)a3.w;
    return r;
}

// Build an A fragment from a row-major bf16 row (this lane's row).
__device__ inline v16bf load_a_bf16(const bf16* __restrict__ row, int kbase, int half) {
    const bf16* p0 = row + kbase + (half ? 8 : 0);
    bf16x8 c0 = *(const bf16x8*)(p0);
    bf16x8 c1 = *(const bf16x8*)(p0 + 16);
    v16bf r;
#pragma unroll
    for (int i = 0; i < 8; ++i) { r[i] = c0[i]; r[8 + i] = c1[i]; }
    return r;
}

// ---------------------------------------------------------------------------
// Async copy of one 16B chunk of a B-fragment stage into LDS.
// A stage = 4 fragments (i,f,g,o gate tiles at this nt) = 4 KB.
// Wave wv (0..7) moves fragment g = wv>>1, half = wv&1: one wave-wide
// global_load_async_to_lds_b128 (32 lanes x 16B = 512B), tracked by ASYNCcnt.
// ---------------------------------------------------------------------------
__device__ inline void issue_b_stage(const bf16* __restrict__ Wc, int kt, int nt,
                                     int wv, int lane, bf16* dstStage) {
    const int g    = wv >> 1;
    const int part = wv & 1;
    size_t fi = (size_t)kt * NT_COMB + (size_t)(g * 32 + nt);
    const bf16* src = Wc + fi * 512 + (size_t)lane * 16 + part * 8;
    bf16*       dst = dstStage + g * 512 + lane * 16 + part * 8;
    unsigned lds_off = (unsigned)(size_t)dst;   // flat LDS addr low 32 bits = LDS byte offset
    asm volatile("global_load_async_to_lds_b128 %0, %1, off"
                 :: "v"(lds_off), "v"(src)
                 : "memory");
}

// ---------------------------------------------------------------------------
// Pack [W_ih; W_hh] (row-major (4H, E/H) fp32) into bf16 B-fragment order:
// fragment fi = kt*NT_COMB + ntg, layout [fi][lane][16 elems] (32B per lane).
// ---------------------------------------------------------------------------
__global__ void pack_comb_kernel(const float* __restrict__ Wih,
                                 const float* __restrict__ Whh,
                                 bf16* __restrict__ Wc) {
    int idx = blockIdx.x * blockDim.x + threadIdx.x;
    if (idx >= KCOMB * NCOMB) return;
    int frag   = idx >> 9;        // /512 elems per fragment
    int within = idx & 511;
    int lane   = within >> 4;
    int e      = within & 15;
    int v = e >> 1, p = e & 1;
    int kt  = frag >> 7;          // / NT_COMB
    int ntg = frag & (NT_COMB - 1);
    int n    = ntg * 16 + (lane & 15);
    int half = lane >> 4;
    int k    = kt * 32 + kmap(v, half, p);
    float val = (k < EMBED) ? Wih[(size_t)n * EMBED + k]
                            : Whh[(size_t)n * HIDDEN + (k - EMBED)];
    Wc[idx] = (bf16)val;
}

// Pack W_fc (OUTD, HIDDEN) fp32 -> bf16 fragments, N padded to 112 with zeros.
__global__ void pack_fc_kernel(const float* __restrict__ Wfc, bf16* __restrict__ Wp) {
    int idx = blockIdx.x * blockDim.x + threadIdx.x;
    if (idx >= KT_FC * NT_FC * 512) return;
    int frag   = idx >> 9;
    int within = idx & 511;
    int lane   = within >> 4;
    int e      = within & 15;
    int v = e >> 1, p = e & 1;
    int kt = frag / NT_FC;
    int nt = frag % NT_FC;
    int n    = nt * 16 + (lane & 15);
    int half = lane >> 4;
    int k    = kt * 32 + kmap(v, half, p);
    float val = (n < OUTD) ? Wfc[(size_t)n * HIDDEN + k] : 0.0f;
    Wp[idx] = (bf16)val;
}

// ---------------------------------------------------------------------------
// FC body: out[:, tcur, :] = sigmoid(h @ W_fc^T + b_fc), one (mt,nt) tile/wave.
// ---------------------------------------------------------------------------
__device__ inline void fc_tile(const bf16*  __restrict__ h,
                               const bf16*  __restrict__ Wp,
                               const float* __restrict__ bfc,
                               float*       __restrict__ out,
                               int w, int lane, int tcur) {
    const int mt   = w / NT_FC;
    const int nt   = w % NT_FC;
    const int half = lane >> 4;
    const int ml   = lane & 15;

    v8f acc = {};
    const bf16* hrow = h + (size_t)(mt * 16 + ml) * HIDDEN;

    for (int kt = 0; kt < KT_FC; ++kt) {
        v16bf a = load_a_bf16(hrow, kt * 32, half);
        v16bf b = *(const v16bf*)(Wp + ((size_t)(kt * NT_FC + nt)) * 512 + (size_t)lane * 16);
        acc = __builtin_amdgcn_wmma_f32_16x16x32_bf16(false, a, false, b, (short)0, acc, false, false);
    }

    const int n = nt * 16 + ml;
    if (n < OUTD) {
        const float bb = bfc[n];
#pragma unroll
        for (int v = 0; v < 8; ++v) {
            int mrow = mt * 16 + v + half * 8;
            if (mrow < BATCH)
                out[((size_t)mrow * TSTEPS + tcur) * OUTD + n] = sigmoidf(acc[v] + bb);
        }
    }
}

// ---------------------------------------------------------------------------
// One LSTM timestep, fused with the FC of the previous step:
//   blocks 0..127   : gates = [x_t, h_prev] @ Wcomb + b -> cell update -> hnext
//   blocks 128..155 : out[:, t-1, :] = sigmoid(hprev @ W_fc^T + b_fc)  (t > 0)
// Cell: all 8 waves of a block share nt, so the 4 gate B-fragments per K-tile
// are staged in LDS: 4 buffers, async-DMA prefetch 2 stages ahead, a single
// workgroup barrier per stage.
// ---------------------------------------------------------------------------
__global__ void __launch_bounds__(256)
lstm_step_kernel(const float* __restrict__ x,      // (BATCH, T*EMBED)
                 const bf16*  __restrict__ Wc,     // packed cell fragments
                 const float* __restrict__ bih,
                 const float* __restrict__ bhh,
                 const bf16*  __restrict__ hprev,  // (MPAD, HIDDEN) bf16
                 bf16*        __restrict__ hnext,  // (MPAD, HIDDEN) bf16
                 float*       __restrict__ c,      // (MPAD, HIDDEN) f32
                 const bf16*  __restrict__ Wp,     // packed FC fragments
                 const float* __restrict__ bfc,
                 float*       __restrict__ out,    // (BATCH, T, OUTD)
                 int t) {
    __shared__ bf16 smem[4][4 * 512];              // 4 stages x 4 fragments x 1KB = 16KB

    const int lane = threadIdx.x & 31;
    const int wv   = threadIdx.x >> 5;             // wave in block, 0..7

    if (blockIdx.x >= CELL_BLOCKS) {
        // ---- fused FC for the previous timestep ----
        if (t > 0)
            fc_tile(hprev, Wp, bfc, out,
                    ((int)blockIdx.x - CELL_BLOCKS) * 8 + wv, lane, t - 1);
        return;
    }

    const int w    = (blockIdx.x << 3) + wv;                   // 0..1023
    const int mt   = w & 31;
    const int nt   = w >> 5;                                   // 0..31 (same for whole block)
    const int half = lane >> 4;
    const int ml   = lane & 15;

    v8f acc0 = {}, acc1 = {}, acc2 = {}, acc3 = {};

    const int m = mt * 16 + ml;                 // this lane's A row
    const int mclamp = (m < BATCH) ? m : (BATCH - 1);
    const float* xrow = x + (size_t)mclamp * (TSTEPS * EMBED) + (size_t)t * EMBED;
    const bf16*  hrow = hprev + (size_t)m * HIDDEN;

    // Prologue: prefetch stages 0 and 1.
    issue_b_stage(Wc, 0, nt, wv, lane, smem[0]);
    issue_b_stage(Wc, 1, nt, wv, lane, smem[1]);

#define LSTM_STAGE(KT, AFRAG)                                                        \
    {                                                                                \
        if ((KT) + 2 < KT_COMB) {                                                    \
            issue_b_stage(Wc, (KT) + 2, nt, wv, lane, smem[((KT) + 2) & 3]);         \
            asm volatile("s_wait_asynccnt 0x2" ::: "memory");                        \
        } else if ((KT) + 1 < KT_COMB) {                                             \
            asm volatile("s_wait_asynccnt 0x1" ::: "memory");                        \
        } else {                                                                     \
            asm volatile("s_wait_asynccnt 0x0" ::: "memory");                        \
        }                                                                            \
        __syncthreads(); /* stage KT visible; also fences buffer reuse (4-deep) */   \
        const bf16* sb = &smem[(KT) & 3][0] + (size_t)lane * 16;                     \
        v16bf b0 = *(const v16bf*)(sb + 0 * 512);                                    \
        v16bf b1 = *(const v16bf*)(sb + 1 * 512);                                    \
        v16bf b2 = *(const v16bf*)(sb + 2 * 512);                                    \
        v16bf b3 = *(const v16bf*)(sb + 3 * 512);                                    \
        acc0 = __builtin_amdgcn_wmma_f32_16x16x32_bf16(false, (AFRAG), false, b0,    \
                                                       (short)0, acc0, false, false);\
        acc1 = __builtin_amdgcn_wmma_f32_16x16x32_bf16(false, (AFRAG), false, b1,    \
                                                       (short)0, acc1, false, false);\
        acc2 = __builtin_amdgcn_wmma_f32_16x16x32_bf16(false, (AFRAG), false, b2,    \
                                                       (short)0, acc2, false, false);\
        acc3 = __builtin_amdgcn_wmma_f32_16x16x32_bf16(false, (AFRAG), false, b3,    \
                                                       (short)0, acc3, false, false);\
    }

    // Phase 1: K = 0..511 comes from x_t (fp32 -> bf16 in registers).
#pragma unroll 1
    for (int kt = 0; kt < 16; ++kt) {
        v16bf a = load_a_f32(xrow, kt * 32, half);
        LSTM_STAGE(kt, a)
    }
    // Phase 2: K = 512..1023 comes from h_prev (bf16).
#pragma unroll 1
    for (int kt = 16; kt < 32; ++kt) {
        v16bf a = load_a_bf16(hrow, (kt - 16) * 32, half);
        LSTM_STAGE(kt, a)
    }
#undef LSTM_STAGE

    // Pointwise cell update.  C/D layout: col n = lane%16, row = v + 8*half.
    const int n = nt * 16 + ml;
    const float bi = bih[n]              + bhh[n];
    const float bf = bih[HIDDEN + n]     + bhh[HIDDEN + n];
    const float bg = bih[2 * HIDDEN + n] + bhh[2 * HIDDEN + n];
    const float bo = bih[3 * HIDDEN + n] + bhh[3 * HIDDEN + n];

#pragma unroll
    for (int v = 0; v < 8; ++v) {
        int mrow = mt * 16 + v + half * 8;
        float gi = sigmoidf(acc0[v] + bi);
        float gf = sigmoidf(acc1[v] + bf);
        float gg = tanhf(acc2[v] + bg);
        float go = sigmoidf(acc3[v] + bo);
        size_t ix = (size_t)mrow * HIDDEN + n;
        float cn = gf * c[ix] + gi * gg;
        c[ix] = cn;
        hnext[ix] = (bf16)(go * tanhf(cn));
    }
}

// Standalone FC kernel for the final timestep's output.
__global__ void __launch_bounds__(256)
fc_step_kernel(const bf16*  __restrict__ h,
               const bf16*  __restrict__ Wp,
               const float* __restrict__ bfc,
               float*       __restrict__ out,
               int t) {
    const int lane = threadIdx.x & 31;
    const int w    = (blockIdx.x << 3) + (threadIdx.x >> 5);   // 0..223
    fc_tile(h, Wp, bfc, out, w, lane, t);
}

// ---------------------------------------------------------------------------
// Host-side launch.
// ---------------------------------------------------------------------------
extern "C" void kernel_launch(void* const* d_in, const int* in_sizes, int n_in,
                              void* d_out, int out_size, void* d_ws, size_t ws_size,
                              hipStream_t stream) {
    const float* inp  = (const float*)d_in[0];
    const float* Wih  = (const float*)d_in[1];
    const float* Whh  = (const float*)d_in[2];
    const float* bih  = (const float*)d_in[3];
    const float* bhh  = (const float*)d_in[4];
    const float* Wfc  = (const float*)d_in[5];
    const float* bfc  = (const float*)d_in[6];
    float* out = (float*)d_out;

    // Workspace layout
    char* ws = (char*)d_ws;
    bf16*  Wc  = (bf16*)(ws);                                   // 4 MiB
    bf16*  Wp  = (bf16*)(ws + (size_t)KCOMB * NCOMB * 2);       // 112 KiB
    char*  p   = ws + (size_t)KCOMB * NCOMB * 2 + (size_t)KT_FC * NT_FC * 512 * 2;
    bf16*  h0  = (bf16*)(p);                                    // 512 KiB
    bf16*  h1  = (bf16*)(p + (size_t)MPAD * HIDDEN * 2);        // 512 KiB
    float* cbuf = (float*)(p + 2 * (size_t)MPAD * HIDDEN * 2);  // 1 MiB

    // h0 = 0, c = 0 each call (deterministic).
    hipMemsetAsync(h0,   0, (size_t)MPAD * HIDDEN * sizeof(bf16),  stream);
    hipMemsetAsync(cbuf, 0, (size_t)MPAD * HIDDEN * sizeof(float), stream);

    // One-time weight packing (cheap; re-done each call for determinism).
    {
        int total = KCOMB * NCOMB;
        pack_comb_kernel<<<(total + 255) / 256, 256, 0, stream>>>(Wih, Whh, Wc);
        int totalf = KT_FC * NT_FC * 512;
        pack_fc_kernel<<<(totalf + 255) / 256, 256, 0, stream>>>(Wfc, Wp);
    }

    // Recurrent scan: cell for step t fused with FC for step t-1.
    for (int t = 0; t < TSTEPS; ++t) {
        const bf16* hp = (t & 1) ? h1 : h0;
        bf16*       hn = (t & 1) ? h0 : h1;
        int grid = (t == 0) ? CELL_BLOCKS : (CELL_BLOCKS + FC_BLOCKS);
        lstm_step_kernel<<<grid, 256, 0, stream>>>(inp, Wc, bih, bhh, hp, hn, cbuf,
                                                   Wp, bfc, out, t);
    }
    // FC for the final timestep (h after step 255 lives in h0: 255 is odd).
    fc_step_kernel<<<FC_BLOCKS, 256, 0, stream>>>(h0, Wp, bfc, out, TSTEPS - 1);
}
